// GNNModel_36945308680549
// MI455X (gfx1250) — matrix operand order repaired and checked by reading, not compiled, not run on hardware
//
#include <hip/hip_runtime.h>
#include <hip/hip_bf16.h>

// ---------------------------------------------------------------------------
// GCN (3x GCNConv + LN + LeakyReLU) -> mean pool -> FFN, for gfx1250 (MI455X).
// Core GEMMs [20000,1024]x[1024,1024] on v_wmma_f32_16x16x32_bf16 with
// triple-buffered LDS tiles filled by GLOBAL_LOAD_ASYNC_TO_LDS_B128 and
// s_wait_asynccnt 3 so the next tile's copy stays in flight across compute.
// Layer 1 (IN=1) is rank-1 and collapses to a scalar segment-sum.
// ---------------------------------------------------------------------------

#define NNODES 20000
#define NEDGES 160000
#define HDIM   1024
#define NGRAPH 16
#define EPSLN  1e-5f
#define SLOPE  0.01f

typedef __attribute__((ext_vector_type(16))) __bf16 v16bf;
typedef __attribute__((ext_vector_type(8)))  float  v8f;
typedef __attribute__((ext_vector_type(4)))  float  f32x4;
typedef __attribute__((ext_vector_type(4)))  int    i32x4;

union Frag { v16bf v; f32x4 f4[2]; };

__device__ __forceinline__ unsigned short f2bf(float x) {
  unsigned int u = __float_as_uint(x);
  u += 0x7FFFu + ((u >> 16) & 1u);        // round to nearest even
  return (unsigned short)(u >> 16);
}
__device__ __forceinline__ float bf2f(unsigned short h) {
  return __uint_as_float(((unsigned int)h) << 16);
}

// ---- CDNA5 async global->LDS path (probe via __has_builtin) ---------------
#if defined(__has_builtin)
#if __has_builtin(__builtin_amdgcn_global_load_async_to_lds_b128)
#define HAVE_ASYNC 1
#endif
#endif
#ifndef HAVE_ASYNC
#define HAVE_ASYNC 0
#endif

#if HAVE_ASYNC
// builtin signature (probe-verified): (int4 AS(1)*, int4 AS(3)*, imm, imm)
typedef __attribute__((address_space(1))) i32x4* gas1p;
typedef __attribute__((address_space(3))) i32x4* las3p;
__device__ __forceinline__ void async_cp16(const void* g, void* l) {
  __builtin_amdgcn_global_load_async_to_lds_b128((gas1p)(void*)g, (las3p)l, 0, 0);
}
#endif

__device__ __forceinline__ void wait_async0() {
#if defined(__has_builtin)
#if __has_builtin(__builtin_amdgcn_s_wait_asynccnt)
  __builtin_amdgcn_s_wait_asynccnt(0);
  return;
#endif
#endif
  asm volatile("s_wait_asynccnt 0x0" ::: "memory");
}
__device__ __forceinline__ void wait_async3() {
#if defined(__has_builtin)
#if __has_builtin(__builtin_amdgcn_s_wait_asynccnt)
  __builtin_amdgcn_s_wait_asynccnt(3);
  return;
#endif
#endif
  asm volatile("s_wait_asynccnt 0x3" ::: "memory");
}

// ---------------------------------------------------------------- init ------
__global__ void k_init(float* __restrict__ deg, float* __restrict__ pooled,
                       float* __restrict__ cnt) {
  int i = blockIdx.x * 256 + threadIdx.x;
  if (i < NNODES) deg[i] = 1.0f;                 // self-loop contributes 1
  if (i < NGRAPH * HDIM) pooled[i] = 0.0f;
  if (i < NGRAPH) cnt[i] = 0.0f;
}

// ------------------------------------------------------------- degrees ------
__global__ void k_deg(const int* __restrict__ dst, float* __restrict__ deg) {
  int e = blockIdx.x * 256 + threadIdx.x;
  if (e < NEDGES) atomicAdd(&deg[dst[e]], 1.0f);
}

__global__ void k_dinv(const float* __restrict__ deg, const float* __restrict__ x,
                       float* __restrict__ dinv, float* __restrict__ s1) {
  int n = blockIdx.x * 256 + threadIdx.x;
  if (n < NNODES) {
    float di = rsqrtf(deg[n]);                   // deg >= 1 always
    dinv[n] = di;
    s1[n] = di * di * x[n];                      // self-loop term of layer 1
  }
}

// norm[e] and layer-1 scalar aggregation fused (layer 1 is rank-1: IN==1)
__global__ void k_norm_l1(const int* __restrict__ src, const int* __restrict__ dst,
                          const float* __restrict__ dinv, const float* __restrict__ x,
                          float* __restrict__ norm, float* __restrict__ s1) {
  int e = blockIdx.x * 256 + threadIdx.x;
  if (e < NEDGES) {
    int s = src[e], d = dst[e];
    float w = dinv[s] * dinv[d];
    norm[e] = w;
    atomicAdd(&s1[d], w * x[s]);
  }
}

// ----------------------------------------------- LayerNorm + LeakyReLU ------
// Layer 1: row = s1[n]*W1 + b1, then LN(g,be) + act, emit bf16.
__global__ __launch_bounds__(256) void k_l1_ln_act(
    const float* __restrict__ s1, const float* __restrict__ W1,
    const float* __restrict__ b1, const float* __restrict__ g,
    const float* __restrict__ be, unsigned short* __restrict__ hbf) {
  __shared__ float sred[256];
  __shared__ float qred[256];
  const int n = blockIdx.x, t = threadIdx.x, f0 = t * 4;
  const float sv = s1[n];
  const f32x4 w = *(const f32x4*)(W1 + f0);
  const f32x4 bb = *(const f32x4*)(b1 + f0);
  f32x4 vals = w * sv + bb;
  float lsum = vals[0] + vals[1] + vals[2] + vals[3];
  float lsq  = vals[0]*vals[0] + vals[1]*vals[1] + vals[2]*vals[2] + vals[3]*vals[3];
  sred[t] = lsum; qred[t] = lsq; __syncthreads();
  for (int s = 128; s > 0; s >>= 1) {
    if (t < s) { sred[t] += sred[t + s]; qred[t] += qred[t + s]; }
    __syncthreads();
  }
  const float mu  = sred[0] * (1.0f / HDIM);
  const float var = qred[0] * (1.0f / HDIM) - mu * mu;
  const float inv = rsqrtf(var + EPSLN);
  const f32x4 gg = *(const f32x4*)(g + f0);
  const f32x4 ee = *(const f32x4*)(be + f0);
#pragma unroll
  for (int i = 0; i < 4; ++i) {
    float y = (vals[i] - mu) * inv * gg[i] + ee[i];
    y = (y >= 0.f) ? y : SLOPE * y;
    hbf[(size_t)n * HDIM + f0 + i] = f2bf(y);
  }
}

// Generic: row = agg[n,:], LN + act, emit bf16.
__global__ __launch_bounds__(256) void k_ln_act(
    const float* __restrict__ in, const float* __restrict__ g,
    const float* __restrict__ be, unsigned short* __restrict__ hbf) {
  __shared__ float sred[256];
  __shared__ float qred[256];
  const int n = blockIdx.x, t = threadIdx.x, f0 = t * 4;
  f32x4 vals = *(const f32x4*)(in + (size_t)n * HDIM + f0);
  float lsum = vals[0] + vals[1] + vals[2] + vals[3];
  float lsq  = vals[0]*vals[0] + vals[1]*vals[1] + vals[2]*vals[2] + vals[3]*vals[3];
  sred[t] = lsum; qred[t] = lsq; __syncthreads();
  for (int s = 128; s > 0; s >>= 1) {
    if (t < s) { sred[t] += sred[t + s]; qred[t] += qred[t + s]; }
    __syncthreads();
  }
  const float mu  = sred[0] * (1.0f / HDIM);
  const float var = qred[0] * (1.0f / HDIM) - mu * mu;
  const float inv = rsqrtf(var + EPSLN);
  const f32x4 gg = *(const f32x4*)(g + f0);
  const f32x4 ee = *(const f32x4*)(be + f0);
#pragma unroll
  for (int i = 0; i < 4; ++i) {
    float y = (vals[i] - mu) * inv * gg[i] + ee[i];
    y = (y >= 0.f) ? y : SLOPE * y;
    hbf[(size_t)n * HDIM + f0 + i] = f2bf(y);
  }
}

// --------------------------------------------- weight transpose -> bf16 -----
// Wt[n,k] = bf16(W[k,n]) via LDS 32x33 tile; both sides coalesced.
__global__ __launch_bounds__(256) void k_wt(const float* __restrict__ W,
                                            unsigned short* __restrict__ Wt) {
  __shared__ float tile[32][33];
  const int k0 = blockIdx.x * 32, n0 = blockIdx.y * 32;
  const int cx = threadIdx.x & 31, ry = threadIdx.x >> 5;   // 32 x 8
#pragma unroll
  for (int i = 0; i < 4; ++i) {
    int r = ry + i * 8;
    tile[r][cx] = W[(size_t)(k0 + r) * HDIM + n0 + cx];
  }
  __syncthreads();
#pragma unroll
  for (int i = 0; i < 4; ++i) {
    int r = ry + i * 8;
    Wt[(size_t)(n0 + r) * HDIM + k0 + cx] = f2bf(tile[cx][r]);
  }
}

// ------------------------------------------------------- WMMA GEMM ----------
// C[M,N] = A[M,K] * Bt[N,K]^T, M=20000, N=K=1024, bf16 in, f32 accum.
// Block: 256 threads = 8 waves (2M x 4N), block tile 32x64, K-step 64.
// Triple-buffered LDS; each thread issues 3 async b128 copies per tile.
// Pipeline: issue tile(i+1) -> s_wait_asynccnt 3 (tile i done, i+1 in flight)
// -> barrier -> 2 wmma from tile i.  Async completions are in-order, so the
// threshold-3 wait retires exactly the current tile's copies.
#define KSTEP 64
#define NTILE (HDIM / KSTEP)
#define LDK   72   // padded LDS row stride in bf16 elems (144 B = 9 x 16 B)
__global__ __launch_bounds__(256) void k_gemm(
    const unsigned short* __restrict__ A, const unsigned short* __restrict__ Bt,
    float* __restrict__ C) {
  __shared__ __align__(16) unsigned short As[3][32 * LDK];
  __shared__ __align__(16) unsigned short Bs[3][64 * LDK];
  const int tid  = threadIdx.x;
  const int lane = tid & 31, wid = tid >> 5;
  const int wm = wid >> 2, wn = wid & 3;
  const int m0 = blockIdx.y * 32;
  const int n0 = blockIdx.x * 64;
  const int half = lane >> 4, lr = lane & 15;

  // cooperative chunk map: 16-byte chunks, 8 per 128-B row slice
  const int crow = tid >> 3, cc = (tid & 7) * 16;           // cc in bytes
  const size_t aoff  = ((size_t)(m0 + crow) * HDIM) * 2 + cc;
  const size_t boff0 = ((size_t)(n0 + crow) * HDIM) * 2 + cc;
  const size_t boff1 = ((size_t)(n0 + 32 + crow) * HDIM) * 2 + cc;
  char* lA  = (char*)&As[0][crow * LDK] + cc;
  char* lB0 = (char*)&Bs[0][crow * LDK] + cc;
  char* lB1 = (char*)&Bs[0][(32 + crow) * LDK] + cc;
  const size_t bufA = sizeof(As[0]);          // byte stride between buffers
  const size_t bufB = sizeof(Bs[0]);

  v8f c = {};
#if HAVE_ASYNC
  // prologue: tile 0 into buffer 0
  async_cp16((const char*)A + aoff, lA);
  async_cp16((const char*)Bt + boff0, lB0);
  async_cp16((const char*)Bt + boff1, lB1);
#pragma unroll
  for (int i = 0; i < NTILE; ++i) {
    if (i + 1 < NTILE) {
      // Issue tile i+1 into buf (i+1)%3 == buf (i-2)%3: its last readers
      // (compute i-2) are past barrier i-1, so the write cannot race.
      const size_t kb = (size_t)(i + 1) * KSTEP * 2;        // bytes
      const size_t pb = (size_t)((i + 1) % 3);
      async_cp16((const char*)A + aoff + kb, lA + pb * bufA);
      async_cp16((const char*)Bt + boff0 + kb, lB0 + pb * bufB);
      async_cp16((const char*)Bt + boff1 + kb, lB1 + pb * bufB);
      wait_async3();        // oldest 3 (tile i) done; tile i+1 stays in flight
    } else {
      wait_async0();        // final tile: drain everything
    }
    __syncthreads();        // everyone's tile-i data visible in LDS
    const unsigned short* as = &As[i % 3][(wm * 16 + lr) * LDK];
    const unsigned short* bs = &Bs[i % 3][(wn * 16 + lr) * LDK];
#pragma unroll
    for (int ks = 0; ks < KSTEP; ks += 32) {
      Frag a, b;
      const unsigned short* ap = as + ks + 8 * half;   // A: K in {8h..}U{16+8h..}
      a.f4[0] = *(const f32x4*)(ap);
      a.f4[1] = *(const f32x4*)(ap + 16);
      const unsigned short* bp = bs + ks + 8 * half;   // B: K in {8h..}U{16+8h..}
      b.f4[0] = *(const f32x4*)(bp);
      b.f4[1] = *(const f32x4*)(bp + 16);
      c = __builtin_amdgcn_wmma_f32_16x16x32_bf16(
          false, a.v, false, b.v, (short)0, c, false, false);
    }
  }
#else
  // fallback: single buffer, b128 VGPR staging
  for (int kb = 0; kb < HDIM; kb += KSTEP) {
    __syncthreads();
    f32x4 ra  = *(const f32x4*)((const char*)A  + aoff  + (size_t)kb * 2);
    f32x4 rb0 = *(const f32x4*)((const char*)Bt + boff0 + (size_t)kb * 2);
    f32x4 rb1 = *(const f32x4*)((const char*)Bt + boff1 + (size_t)kb * 2);
    *(f32x4*)lA  = ra;
    *(f32x4*)lB0 = rb0;
    *(f32x4*)lB1 = rb1;
    __syncthreads();
    const unsigned short* as = &As[0][(wm * 16 + lr) * LDK];
    const unsigned short* bs = &Bs[0][(wn * 16 + lr) * LDK];
#pragma unroll
    for (int ks = 0; ks < KSTEP; ks += 32) {
      Frag a, b;
      const unsigned short* ap = as + ks + 8 * half;
      a.f4[0] = *(const f32x4*)(ap);
      a.f4[1] = *(const f32x4*)(ap + 16);
      const unsigned short* bp = bs + ks + 8 * half;
      b.f4[0] = *(const f32x4*)(bp);
      b.f4[1] = *(const f32x4*)(bp + 16);
      c = __builtin_amdgcn_wmma_f32_16x16x32_bf16(
          false, a.v, false, b.v, (short)0, c, false, false);
    }
  }
#endif

  // C layout: VGPR r, lanes 0-15 -> M=r, lanes 16-31 -> M=8+r; N = lane&15
#pragma unroll
  for (int r = 0; r < 8; ++r) {
    int m = m0 + wm * 16 + half * 8 + r;
    int n = n0 + wn * 16 + lr;
    C[(size_t)m * HDIM + n] = c[r];
  }
}

// ------------------------------------------------- message aggregation ------
// agg[n,:] = dinv[n]^2 * hw[n,:]  (self-loop; also initializes agg)
__global__ __launch_bounds__(256) void k_agg_init(
    const float* __restrict__ hw, const float* __restrict__ dinv,
    float* __restrict__ agg) {
  const int n = blockIdx.x;
  const int f0 = threadIdx.x * 4;
  const float di = dinv[n];
  f32x4 v = *(const f32x4*)(hw + (size_t)n * HDIM + f0);
  *(f32x4*)(agg + (size_t)n * HDIM + f0) = v * (di * di);
}

// agg[dst,:] += norm[e] * hw[src,:]  (one block per edge, b128 loads)
__global__ __launch_bounds__(256) void k_agg_edges(
    const int* __restrict__ src, const int* __restrict__ dst,
    const float* __restrict__ norm, const float* __restrict__ hw,
    float* __restrict__ agg) {
  const int e = blockIdx.x;
  const int s = src[e], d = dst[e];
  const float w = norm[e];
  const int f0 = threadIdx.x * 4;
  f32x4 v = *(const f32x4*)(hw + (size_t)s * HDIM + f0) * w;
  float* ad = agg + (size_t)d * HDIM + f0;
  atomicAdd(&ad[0], v[0]);
  atomicAdd(&ad[1], v[1]);
  atomicAdd(&ad[2], v[2]);
  atomicAdd(&ad[3], v[3]);
}

// ---------------------------------------------------------------- pool ------
__global__ __launch_bounds__(256) void k_pool(
    const unsigned short* __restrict__ hbf, const int* __restrict__ batch,
    float* __restrict__ pooled, float* __restrict__ cnt) {
  const int n = blockIdx.x;
  const int b = batch[n];
  if (threadIdx.x == 0) atomicAdd(&cnt[b], 1.0f);
  for (int f = threadIdx.x; f < HDIM; f += 256)
    atomicAdd(&pooled[(size_t)b * HDIM + f], bf2f(hbf[(size_t)n * HDIM + f]));
}

// ----------------------------------------------------------------- FFN ------
// out[g] = ((pooled[g]/cnt[g]) @ fw1 + fb1) @ fw2 + fb2   (16x1024x256x1)
__global__ __launch_bounds__(256) void k_ffn(
    const float* __restrict__ pooled, const float* __restrict__ cnt,
    const float* __restrict__ fw1, const float* __restrict__ fb1,
    const float* __restrict__ fw2, const float* __restrict__ fb2,
    float* __restrict__ out) {
  __shared__ float t1[NGRAPH * 256];
  __shared__ float red[256];
  const int j = threadIdx.x;
  for (int g = 0; g < NGRAPH; ++g) {
    const float inv = 1.0f / fmaxf(cnt[g], 1.0f);
    const float* pr = pooled + (size_t)g * HDIM;
    float acc = 0.f;
    for (int k = 0; k < HDIM; ++k) acc += pr[k] * fw1[(size_t)k * 256 + j];
    t1[g * 256 + j] = acc * inv + fb1[j];
  }
  __syncthreads();
  const float w2 = fw2[j];
  for (int g = 0; g < NGRAPH; ++g) {
    red[j] = t1[g * 256 + j] * w2;
    __syncthreads();
    for (int s = 128; s > 0; s >>= 1) {
      if (j < s) red[j] += red[j + s];
      __syncthreads();
    }
    if (j == 0) out[g] = red[0] + fb2[0];
    __syncthreads();
  }
}

// ---------------------------------------------------------------------------
extern "C" void kernel_launch(void* const* d_in, const int* in_sizes, int n_in,
                              void* d_out, int out_size, void* d_ws, size_t ws_size,
                              hipStream_t stream) {
  (void)in_sizes; (void)n_in; (void)out_size; (void)ws_size;
  const float* x    = (const float*)d_in[0];
  const int*   edge = (const int*)d_in[1];        // [2,E]: src = edge, dst = edge+E
  const int*   batch= (const int*)d_in[2];
  const float* W1   = (const float*)d_in[3];
  const float* b1   = (const float*)d_in[4];
  const float* W2   = (const float*)d_in[5];
  const float* W3   = (const float*)d_in[7];
  const float* g1   = (const float*)d_in[9];
  const float* be1  = (const float*)d_in[10];
  const float* g2   = (const float*)d_in[11];
  const float* be2  = (const float*)d_in[12];
  const float* g3   = (const float*)d_in[13];
  const float* be3  = (const float*)d_in[14];
  const float* fw1  = (const float*)d_in[15];
  const float* fb1  = (const float*)d_in[16];
  const float* fw2  = (const float*)d_in[17];
  const float* fb2  = (const float*)d_in[18];
  const int* src = edge;
  const int* dst = edge + NEDGES;

  // ---- workspace layout (all chunks 16B-aligned) ----
  char* p = (char*)d_ws;
  float* dinv = (float*)p;                    p += (size_t)NNODES * 4;
  float* deg  = (float*)p;                    p += (size_t)NNODES * 4;
  float* norm = (float*)p;                    p += (size_t)NEDGES * 4;
  float* s1   = (float*)p;                    p += (size_t)NNODES * 4;
  unsigned short* hbf = (unsigned short*)p;   p += (size_t)NNODES * HDIM * 2;
  unsigned short* Wt  = (unsigned short*)p;   p += (size_t)HDIM * HDIM * 2;
  float* hw   = (float*)p;                    p += (size_t)NNODES * HDIM * 4;
  float* agg  = (float*)p;                    p += (size_t)NNODES * HDIM * 4;
  float* pooled = (float*)p;                  p += (size_t)NGRAPH * HDIM * 4;
  float* cnt  = (float*)p;                    p += (size_t)NGRAPH * 4;

  const dim3 blk(256);
  const int nN = (NNODES + 255) / 256;        // 79
  const int nE = (NEDGES + 255) / 256;        // 625

  // degrees / norm / layer-1 scalar aggregation
  k_init   <<<nN, blk, 0, stream>>>(deg, pooled, cnt);
  k_deg    <<<nE, blk, 0, stream>>>(dst, deg);
  k_dinv   <<<nN, blk, 0, stream>>>(deg, x, dinv, s1);
  k_norm_l1<<<nE, blk, 0, stream>>>(src, dst, dinv, x, norm, s1);
  k_l1_ln_act<<<NNODES, blk, 0, stream>>>(s1, W1, b1, g1, be1, hbf);

  const dim3 gWt(HDIM / 32, HDIM / 32);       // (32, 32)
  const dim3 gGemm(HDIM / 64, NNODES / 32);   // (16, 625)

  // layer 2
  k_wt       <<<gWt,   blk, 0, stream>>>(W2, Wt);
  k_gemm     <<<gGemm, blk, 0, stream>>>(hbf, Wt, hw);
  k_agg_init <<<NNODES, blk, 0, stream>>>(hw, dinv, agg);
  k_agg_edges<<<NEDGES, blk, 0, stream>>>(src, dst, norm, hw, agg);
  k_ln_act   <<<NNODES, blk, 0, stream>>>(agg, g2, be2, hbf);

  // layer 3
  k_wt       <<<gWt,   blk, 0, stream>>>(W3, Wt);
  k_gemm     <<<gGemm, blk, 0, stream>>>(hbf, Wt, hw);
  k_agg_init <<<NNODES, blk, 0, stream>>>(hw, dinv, agg);
  k_agg_edges<<<NEDGES, blk, 0, stream>>>(src, dst, norm, hw, agg);
  k_ln_act   <<<NNODES, blk, 0, stream>>>(agg, g3, be3, hbf);

  // pool + FFN
  k_pool<<<NNODES, blk, 0, stream>>>(hbf, batch, pooled, cnt);
  k_ffn <<<1, blk, 0, stream>>>(pooled, cnt, fw1, fb1, fw2, fb2, (float*)d_out);
}
